// BiLSTM_CRF_4183298146614
// MI455X (gfx1250) — matrix operand order repaired and checked by reading, not compiled, not run on hardware
//
#include <hip/hip_runtime.h>
#include <hip/hip_bf16.h>

// ---------------- constants (match reference) ----------------
#define T_LEN   8192
#define HID     512
#define NH      256          // per-direction hidden
#define G4      1024         // 4*NH gate width
#define NTAG    7
#define START_TAG 5
#define STOP_TAG  6
#define NEG_VAL  (-10000.0f)
#define NLAYERS 7

// ---------------- WMMA / TDM types ----------------
typedef __attribute__((ext_vector_type(16))) __bf16        v16bf;
typedef __attribute__((ext_vector_type(8)))  float         v8f;
typedef __attribute__((ext_vector_type(4)))  unsigned int  v4u;
typedef __attribute__((ext_vector_type(8)))  int           v8i_;
typedef __attribute__((ext_vector_type(4)))  int           v4i_;

union FragBF {
    uint4 u[2];   // 32 bytes = 16 bf16
    v16bf v;
};

__device__ __forceinline__ unsigned short f32_to_bf16_rne(float f) {
    unsigned int x = __builtin_bit_cast(unsigned int, f);
    unsigned int lsb = (x >> 16) & 1u;
    x += 0x7FFFu + lsb;
    return (unsigned short)(x >> 16);
}

__device__ __forceinline__ float sigmoidf_fast(float x) {
    return 1.0f / (1.0f + __expf(-x));
}

// ---------------------------------------------------------------------------
// TDM: async 2-D tile load Global -> LDS (Tensor Data Mover, gfx1250).
// D# bit layout per CDNA5 ISA ch.8: group0 {count=1, lds_addr, global_addr,
// type=2}; group1 {data_size=1 (2-byte elems), tensor_dim0/1, tile_dim0/1,
// tensor_dim0_stride}. 2-D tensor => remaining groups zero.
// 6-arg builtin form (clang-23 / therock-10.0 signature).
// ---------------------------------------------------------------------------
__device__ __forceinline__ void tdm_load_2d_bf16(
    unsigned lds_off_bytes, const void* gaddr,
    unsigned tensor_d0, unsigned tensor_d1,
    unsigned tile_d0, unsigned tile_d1,
    unsigned long long stride0_elems)
{
    unsigned long long ga = (unsigned long long)(uintptr_t)gaddr;
    v4u g0;
    g0[0] = 1u;                                            // count=1, user mode
    g0[1] = lds_off_bytes;                                 // lds_addr
    g0[2] = (unsigned)(ga & 0xFFFFFFFFu);                  // global_addr[31:0]
    g0[3] = (unsigned)((ga >> 32) & 0x1FFFFFFu)            // global_addr[56:32]
          | (2u << 30);                                    // type=2 ("image")
    v8i_ g1;
    g1[0] = (int)(1u << 16);                               // data_size=1 (2B), wg_mask=0
    g1[1] = (int)((tensor_d0 & 0xFFFFu) << 16);            // tensor_dim0[15:0]
    g1[2] = (int)((tensor_d0 >> 16) | ((tensor_d1 & 0xFFFFu) << 16));
    g1[3] = (int)((tensor_d1 >> 16) | ((tile_d0 & 0xFFFFu) << 16));
    g1[4] = (int)(tile_d1 & 0xFFFFu);                      // tile_dim1, tile_dim2=0
    g1[5] = (int)(stride0_elems & 0xFFFFFFFFu);            // dim0_stride[31:0]
    g1[6] = (int)((stride0_elems >> 32) & 0xFFFFu);        // dim0_stride[47:32]
    g1[7] = 0;
    v4i_ z4 = {0, 0, 0, 0};
    v8i_ z8 = {0, 0, 0, 0, 0, 0, 0, 0};
    __builtin_amdgcn_tensor_load_to_lds(g0, g1, z4, z4, z8, 0);
}

// ---------------------------------------------------------------------------
// Convert layer input (optionally sum of two skip buffers) fp32 -> bf16,
// padded along K to a multiple of 32 for the 16x16x32 WMMA.
// dst is (T_LEN x kpad) row-major bf16.
// ---------------------------------------------------------------------------
__global__ void cvt_x_kernel(const float* __restrict__ s1,
                             const float* __restrict__ s2,
                             unsigned short* __restrict__ dst,
                             int indim, int kpad)
{
    int idx = blockIdx.x * blockDim.x + threadIdx.x;
    int t = idx / kpad;
    int k = idx - t * kpad;
    if (t >= T_LEN) return;
    float v = 0.0f;
    if (k < indim) {
        v = s1[(size_t)t * indim + k];
        if (s2) v += s2[(size_t)t * indim + k];
    }
    dst[idx] = f32_to_bf16_rne(v);
}

// ---------------------------------------------------------------------------
// Pack Wih (4NH x indim, row-major fp32) into WMMA-B-fragment-native bf16:
//   Bp[kc*32768 + n*32 + kl] = bf16( Wih[n*kreal + (kc*32+kl)] )  (0 if k>=kreal)
// so that a lane's B-fragment (k = 8h+[0..7] and 16+8h+[0..7] of one column) is
// two contiguous 16-byte loads.
// ---------------------------------------------------------------------------
__global__ void cvt_w_kernel(const float* __restrict__ Wih,
                             unsigned short* __restrict__ Bp,
                             int kreal, int kpad)
{
    int idx = blockIdx.x * blockDim.x + threadIdx.x;
    if (idx >= kpad * G4) return;
    int kc  = idx / (G4 * 32);
    int rem = idx - kc * (G4 * 32);
    int n   = rem >> 5;
    int kl  = rem & 31;
    int k   = kc * 32 + kl;
    float v = (k < kreal) ? Wih[(size_t)n * kreal + k] : 0.0f;
    Bp[idx] = f32_to_bf16_rne(v);
}

// ---------------------------------------------------------------------------
// WMMA GEMM: xp(T_LEN x 1024) = X(T_LEN x kpad) @ Bp(kpad x 1024) + bias,
// both directions via blockIdx.z.
// Block = 128 threads (4 waves); block output = 16 rows x 64 cols.
// Wave 0 TDM-loads the 16 x kpad A tile (16KB) and the block's 64-column
// B slab (kpad/32 chunks x 4KB) into LDS, waits on TENSORcnt, then all
// waves assemble fragments with ds_load_b128 and run the wmma K-loop.
// Dynamic LDS: [0,16K) = A tile, [16K, 16K+64K) = B slab.
// ---------------------------------------------------------------------------
__global__ __launch_bounds__(128)
void gemm_xp_kernel(const unsigned short* __restrict__ X,
                    const unsigned short* __restrict__ BpF,
                    const unsigned short* __restrict__ BpR,
                    const float* __restrict__ biasF,
                    const float* __restrict__ biasR,
                    float* __restrict__ xpF,
                    float* __restrict__ xpR,
                    int kpad)
{
    extern __shared__ unsigned short smem_dyn[];   // dyn LDS, offset 0
    unsigned short* As = smem_dyn;                 // 16 x kpad bf16
    unsigned short* Bs = smem_dyn + 8192;          // [kpad/32][2048] bf16

    const int lane = threadIdx.x & 31;
    const int wave = threadIdx.x >> 5;          // 0..3
    const int mt   = blockIdx.x;                // row tile  (0..511)
    const int col0 = blockIdx.y * 64;           // block's first output column

    const unsigned short* Bp  = blockIdx.z ? BpR  : BpF;
    const float*          bia = blockIdx.z ? biasR : biasF;
    float*                Out = blockIdx.z ? xpR  : xpF;

    const int nchunks = kpad >> 5;

    // --- async tile staging via Tensor Data Mover (wave 0 issues) ---
    if (wave == 0) {
        // A tile: rows mt*16 .. mt*16+15, full K
        tdm_load_2d_bf16(0u, X + (size_t)mt * 16 * kpad,
                         (unsigned)kpad, (unsigned)T_LEN,
                         (unsigned)kpad, 16u,
                         (unsigned long long)kpad);
        // B slab: for each k-chunk row (stride 32768 elems), 64 cols * 32 k
        tdm_load_2d_bf16(16384u, Bp + (size_t)col0 * 32,
                         (unsigned)(G4 * 32), (unsigned)nchunks,
                         2048u, (unsigned)nchunks,
                         (unsigned long long)(G4 * 32));
        __builtin_amdgcn_s_wait_tensorcnt(0);
    }
    __syncthreads();

    const int l15  = lane & 15;
    const int hiA  = (lane >> 4) ? 8  : 0;      // A k-offset within chunk
    const int hiB  = (lane >> 4) ? 8  : 0;      // B klocal offset within chunk
    const int lcol = wave * 16 + l15;           // column within block slab

    v8f acc = {};
    const unsigned short* ap = As + l15 * kpad + hiA;
    const unsigned short* bp = Bs + lcol * 32 + hiB;

    for (int kc = 0; kc < nchunks; ++kc) {
        FragBF a, b;
        // A: k = kb+hiA+[0..7] and kb+hiA+[16..23]  (16-bit A VGPR layout)
        a.u[0] = *(const uint4*)(ap);
        a.u[1] = *(const uint4*)(ap + 16);
        // B: k = kb+hiB+[0..7] and kb+hiB+[16..23] for this column (matches A)
        b.u[0] = *(const uint4*)(bp);
        b.u[1] = *(const uint4*)(bp + 16);
        acc = __builtin_amdgcn_wmma_f32_16x16x32_bf16(
                  false, a.v, false, b.v, (short)0, acc, false, false);
        ap += 32;
        bp += 2048;
    }

    // C/D layout: VGPR r -> row mt*16 + r + (lane>=16 ? 8 : 0), col
    const int col   = col0 + lcol;
    const int rbase = mt * 16 + ((lane >> 4) ? 8 : 0);
    const float bc  = bia[col];
#pragma unroll
    for (int r = 0; r < 8; ++r) {
        Out[(size_t)(rbase + r) * G4 + col] = acc[r] + bc;
    }
}

// ---------------------------------------------------------------------------
// Recurrent scan, one workgroup (1024 threads) per direction.
// Thread tid owns gate output tid (gate order i,f,g,o). h broadcast through
// LDS (broadcast reads = conflict-free); Whh streamed from L2 (1MB fp32,
// fully L2-resident). 4 independent accumulator chains -> VOPD dual-issue /
// v_pk_fma_f32 packing instead of one serial FMA chain.
// ---------------------------------------------------------------------------
__global__ __launch_bounds__(1024)
void lstm_scan_kernel(const float* __restrict__ xpF,
                      const float* __restrict__ xpR,
                      const float* __restrict__ WhhF,
                      const float* __restrict__ WhhR,
                      float* __restrict__ out /* T_LEN x HID */)
{
    __shared__ float h_lds[NH];
    __shared__ float gates[G4];

    const int tid = threadIdx.x;
    const int dir = blockIdx.x;                 // 0 = fwd, 1 = bwd
    const float* xp  = dir ? xpR  : xpF;
    const float* Whh = dir ? WhhR : WhhF;
    const int    ofs = dir ? NH : 0;

    if (tid < NH) h_lds[tid] = 0.0f;
    float c = 0.0f;
    __syncthreads();

    const float4* w4 = (const float4*)(Whh + (size_t)tid * NH);

    for (int s = 0; s < T_LEN; ++s) {
        const int t = dir ? (T_LEN - 1 - s) : s;

        // prefetch next step's xp element into cache (global_prefetch_b8)
        if (s + 1 < T_LEN) {
            const int tn = dir ? (t - 1) : (t + 1);
            __builtin_prefetch((const void*)(xp + (size_t)tn * G4 + tid), 0, 3);
        }

        float4 a4 = make_float4(0.f, 0.f, 0.f, 0.f);
        const float4* h4 = (const float4*)h_lds;
#pragma unroll 8
        for (int j = 0; j < NH / 4; ++j) {
            float4 w = w4[j];
            float4 h = h4[j];
            a4.x = __fmaf_rn(w.x, h.x, a4.x);
            a4.y = __fmaf_rn(w.y, h.y, a4.y);
            a4.z = __fmaf_rn(w.z, h.z, a4.z);
            a4.w = __fmaf_rn(w.w, h.w, a4.w);
        }
        float g = xp[(size_t)t * G4 + tid] + ((a4.x + a4.y) + (a4.z + a4.w));
        gates[tid] = g;
        __syncthreads();

        if (tid < NH) {
            float ig = sigmoidf_fast(gates[tid]);
            float fg = sigmoidf_fast(gates[NH + tid]);
            float gg = tanhf(gates[2 * NH + tid]);
            float og = sigmoidf_fast(gates[3 * NH + tid]);
            c = fg * c + ig * gg;
            float h = og * tanhf(c);
            h_lds[tid] = h;
            out[(size_t)t * HID + ofs + tid] = h;
        }
        __syncthreads();
    }
}

// ---------------------------------------------------------------------------
// feats(T_LEN x 7) = o7(T_LEN x 512) @ fc_w.T + fc_b
// ---------------------------------------------------------------------------
__global__ void fc_kernel(const float* __restrict__ o7,
                          const float* __restrict__ fcw,
                          const float* __restrict__ fcb,
                          float* __restrict__ feats)
{
    int idx = blockIdx.x * blockDim.x + threadIdx.x;
    int t   = idx / NTAG;
    int tag = idx - t * NTAG;
    if (t >= T_LEN) return;
    const float4* x = (const float4*)(o7 + (size_t)t * HID);
    const float4* w = (const float4*)(fcw + (size_t)tag * HID);
    float4 a4 = make_float4(0.f, 0.f, 0.f, 0.f);
#pragma unroll 8
    for (int j = 0; j < HID / 4; ++j) {
        float4 a = x[j], b = w[j];
        a4.x = __fmaf_rn(a.x, b.x, a4.x);
        a4.y = __fmaf_rn(a.y, b.y, a4.y);
        a4.z = __fmaf_rn(a.z, b.z, a4.z);
        a4.w = __fmaf_rn(a.w, b.w, a4.w);
    }
    feats[t * NTAG + tag] = fcb[tag] + ((a4.x + a4.y) + (a4.z + a4.w));
}

// ---------------------------------------------------------------------------
// Viterbi decode: single wave. Lanes 0..6 run the forward max-sum recursion,
// lane 0 does termination + backtrace. d_out[0]=score, d_out[1+t]=path[t].
// ---------------------------------------------------------------------------
__global__ void viterbi_kernel(const float* __restrict__ feats,
                               const float* __restrict__ trans,
                               int* __restrict__ bptr,
                               float* __restrict__ outp)
{
    __shared__ float fv[NTAG];
    __shared__ float fvn[NTAG];
    const int n = threadIdx.x;

    if (n < NTAG) fv[n] = (n == START_TAG) ? 0.0f : NEG_VAL;
    __syncthreads();

    for (int t = 0; t < T_LEN; ++t) {
        if (n < NTAG) {
            float best = -3.4e38f;
            int   bi   = 0;
            for (int p = 0; p < NTAG; ++p) {
                float sc = fv[p] + trans[n * NTAG + p];
                if (sc > best) { best = sc; bi = p; }
            }
            fvn[n] = best + feats[t * NTAG + n];
            bptr[t * NTAG + n] = bi;
        }
        __syncthreads();
        if (n < NTAG) fv[n] = fvn[n];
        __syncthreads();
    }

    if (n == 0) {
        float best = -3.4e38f;
        int   bi   = 0;
        for (int p = 0; p < NTAG; ++p) {
            float sc = fv[p] + trans[STOP_TAG * NTAG + p];
            if (sc > best) { best = sc; bi = p; }
        }
        outp[0] = best;
        int tag = bi;
        for (int t = T_LEN - 1; t >= 0; --t) {
            outp[1 + t] = (float)tag;      // path[t]
            tag = bptr[t * NTAG + tag];    // carry into previous step
        }
    }
}

// ---------------------------------------------------------------------------
// Host-side orchestration (all on `stream`, graph-capture safe).
// Input order: sentence, then per layer {fWih,fWhh,fb,rWih,rWhh,rb}, then
// fc_w, fc_b, transitions.
// ---------------------------------------------------------------------------
extern "C" void kernel_launch(void* const* d_in, const int* in_sizes, int n_in,
                              void* d_out, int out_size, void* d_ws, size_t ws_size,
                              hipStream_t stream) {
    (void)in_sizes; (void)n_in; (void)out_size; (void)ws_size;

    const float* sentence = (const float*)d_in[0];
    auto P = [&](int l, int j) -> const float* {
        return (const float*)d_in[1 + 6 * l + j];
    };
    const float* fc_w  = (const float*)d_in[1 + 6 * NLAYERS + 0];
    const float* fc_b  = (const float*)d_in[1 + 6 * NLAYERS + 1];
    const float* trans = (const float*)d_in[1 + 6 * NLAYERS + 2];

    // ---- workspace carve-out (256B aligned bumps) ----
    char* wp = (char*)d_ws;
    auto alloc = [&](size_t bytes) -> char* {
        char* p = wp;
        wp += (bytes + 255) & ~(size_t)255;
        return p;
    };
    float*          xpF  = (float*)alloc((size_t)T_LEN * G4 * 4);   // 32 MB
    float*          xpR  = (float*)alloc((size_t)T_LEN * G4 * 4);   // 32 MB
    float*          oA   = (float*)alloc((size_t)T_LEN * HID * 4);  // 16 MB
    float*          oB   = (float*)alloc((size_t)T_LEN * HID * 4);
    float*          oC   = (float*)alloc((size_t)T_LEN * HID * 4);
    unsigned short* xbf  = (unsigned short*)alloc((size_t)T_LEN * HID * 2); // 8 MB
    unsigned short* WbfF = (unsigned short*)alloc((size_t)HID * G4 * 2);    // 1 MB
    unsigned short* WbfR = (unsigned short*)alloc((size_t)HID * G4 * 2);
    float*          feats = (float*)alloc((size_t)T_LEN * NTAG * 4);
    int*            bptr  = (int*)alloc((size_t)T_LEN * NTAG * 4);

    float* outs[3] = { oA, oB, oC };

    for (int l = 0; l < NLAYERS; ++l) {
        const int indim = (l == 0) ? 3 : HID;
        const int kpad  = (l == 0) ? 32 : HID;

        const float* in1 = (l == 0) ? sentence : outs[(l - 1) % 3];
        const float* in2 = (l >= 2) ? outs[(l - 2) % 3] : nullptr;
        float*       olb = outs[l % 3];

        // 1) input (+skip) -> padded bf16
        {
            int nthr = T_LEN * kpad;
            cvt_x_kernel<<<(nthr + 255) / 256, 256, 0, stream>>>(
                in1, in2, xbf, indim, kpad);
        }
        // 2) weights -> fragment-native bf16 (both directions)
        {
            int nthr = kpad * G4;
            cvt_w_kernel<<<(nthr + 255) / 256, 256, 0, stream>>>(
                P(l, 0), WbfF, indim, kpad);
            cvt_w_kernel<<<(nthr + 255) / 256, 256, 0, stream>>>(
                P(l, 3), WbfR, indim, kpad);
        }
        // 3) WMMA input projection xp = x @ Wih.T + b, both directions
        //    dynamic LDS: 16KB A tile + 64KB B slab
        {
            dim3 grid(T_LEN / 16, (G4 / 16) / 4, 2);
            size_t shmem = 16384 + 65536;
            gemm_xp_kernel<<<grid, 128, shmem, stream>>>(
                xbf, WbfF, WbfR, P(l, 2), P(l, 5), xpF, xpR, kpad);
        }
        // 4) recurrent scan (fwd + bwd), writes (T_LEN x HID) layer output
        lstm_scan_kernel<<<2, 1024, 0, stream>>>(
            xpF, xpR, P(l, 1), P(l, 4), olb);
    }

    // FC projection to tag space
    fc_kernel<<<(T_LEN * NTAG + 255) / 256, 256, 0, stream>>>(
        outs[(NLAYERS - 1) % 3], fc_w, fc_b, feats);

    // Viterbi decode
    viterbi_kernel<<<1, 32, 0, stream>>>(feats, trans, bptr, (float*)d_out);
}
